// WeightedCrossAttention_15702400434593
// MI455X (gfx1250) — matrix operand order repaired and checked by reading, not compiled, not run on hardware
//
#include <hip/hip_runtime.h>
#include <cstdint>
#include <cstddef>

// ---------------------------------------------------------------------------
// WeightedCrossAttention for MI455X (gfx1250, wave32, WMMA, async-to-LDS)
//
//  1) topk_sort   : per-slot hybrid bitonic sort: 16 elems/thread register
//                   sorting-network passes + LDS stages only for j>=16
//                   (128KB LDS/block -- CDNA5's 320KB WGP LDS makes this legal)
//  2) vw_softmax  : softmax over S=256 samples (vw = sm + (1-sm))
//  3) gather_rows : gather features/pos (33.5MB, L2-resident), emit bf16 keys
//                   and vw-scaled values
//  4) gemm_bf16   : WMMA bf16 GEMMs; A tile staged via global_load_async_to_lds
//                   (double buffered, s_wait_asynccnt), B direct (cache-resident)
//  5) attn        : per (slot,head) softmax-attention, wave32 shuffles
//  6) slot_ln     : residual + LayerNorm -> output slots
//  7) seg_scatter : seg softmax + sparse scatter + 3x3 curiosity spread
// ---------------------------------------------------------------------------

#define HW_    16384
#define HH_    128
#define WW_    128
#define NSLOT  2048
#define BATCH  2
#define DIM    256
#define NHEAD  8
#define HD_    32
#define SSAMP  256
#define IMP_   128

typedef __attribute__((ext_vector_type(16))) __bf16 v16bf;
typedef __attribute__((ext_vector_type(8)))  float  v8f;
typedef unsigned long long ull;

// --------------------------- f32 -> bf16 convert ---------------------------
__global__ __launch_bounds__(256)
void cvt_f32_bf16(const float* __restrict__ src, __bf16* __restrict__ dst, int count) {
  int i = blockIdx.x * 256 + threadIdx.x;
  if (i < count) dst[i] = (__bf16)src[i];
}

// --------------------------- per-slot full sort ----------------------------
// key = (~ordered(float) << 32) | idx ; ascending == descending value with
// stable ties on smaller index (matches jnp.argsort(-x)).
// Hybrid bitonic: each thread owns 16 contiguous elements; all j<=8 stages run
// as register sorting networks, only j>=16 stages touch LDS.
__device__ __forceinline__ void cex(ull& a, ull& b, bool up) {
  if ((a > b) == up) { ull t = a; a = b; b = t; }
}

__global__ __launch_bounds__(1024)
void topk_sort(const float* __restrict__ curio, const int* __restrict__ cov_sel,
               int* __restrict__ feat_idx, float* __restrict__ s_cur) {
  extern __shared__ ull keys[];                 // 16384 * 8B = 128 KB
  const int n    = blockIdx.x;
  const int tid  = threadIdx.x;
  const int base = tid * 16;
  const float* cm = curio + (size_t)n * HW_;

  ull r[16];
  #pragma unroll
  for (int l = 0; l < 16; ++l) {
    unsigned u   = __float_as_uint(cm[base + l]);
    unsigned ord = (u & 0x80000000u) ? ~u : (u | 0x80000000u);
    r[l] = ((ull)(~ord) << 32) | (unsigned)(base + l);
  }
  // k = 2,4,8,16 : fully in registers
  #pragma unroll
  for (int kb = 1; kb <= 4; ++kb) {
    const int k = 1 << kb;
    #pragma unroll
    for (int jb = 4; jb >= 0; --jb) {
      const int j = 1 << jb;
      if (j > (k >> 1)) continue;
      #pragma unroll
      for (int l = 0; l < 16; ++l)
        if ((l & j) == 0) cex(r[l], r[l | j], (((base + l) & k) == 0));
    }
  }
  #pragma unroll
  for (int l = 0; l < 16; ++l) keys[base + l] = r[l];

  // k = 32 .. 16384
  for (int k = 32; k <= HW_; k <<= 1) {
    for (int j = k >> 1; j >= 16; j >>= 1) {
      __syncthreads();
      for (int p = tid; p < HW_ / 2; p += 1024) {
        int i   = ((p & ~(j - 1)) << 1) | (p & (j - 1));
        int prt = i + j;
        bool up = ((i & k) == 0);
        ull a = keys[i], b = keys[prt];
        if ((a > b) == up) { keys[i] = b; keys[prt] = a; }
      }
    }
    __syncthreads();
    #pragma unroll
    for (int l = 0; l < 16; ++l) r[l] = keys[base + l];
    const bool up = ((base & k) == 0);          // k>=32: uniform over the 16 elems
    #pragma unroll
    for (int jb = 3; jb >= 0; --jb) {
      const int j = 1 << jb;
      #pragma unroll
      for (int l = 0; l < 16; ++l)
        if ((l & j) == 0) cex(r[l], r[l | j], up);
    }
    #pragma unroll
    for (int l = 0; l < 16; ++l) keys[base + l] = r[l];
  }
  __syncthreads();

  if (tid < SSAMP) {
    int pos = (tid < IMP_) ? tid : (IMP_ + cov_sel[tid - IMP_]);
    int fi  = (int)(keys[pos] & 0xFFFFFFFFull);
    feat_idx[n * SSAMP + tid] = fi;
    s_cur[n * SSAMP + tid]    = cm[fi];
  }
}

// --------------------------- value-weight softmax --------------------------
__global__ __launch_bounds__(256)
void vw_softmax(const float* __restrict__ s_cur, float* __restrict__ vw) {
  __shared__ float red[256];
  const int n = blockIdx.x, tid = threadIdx.x;
  float x = s_cur[n * SSAMP + tid];
  red[tid] = x; __syncthreads();
  for (int o = 128; o > 0; o >>= 1) { if (tid < o) red[tid] = fmaxf(red[tid], red[tid + o]); __syncthreads(); }
  float mx = red[0]; __syncthreads();
  float e = __expf(x - mx);
  red[tid] = e; __syncthreads();
  for (int o = 128; o > 0; o >>= 1) { if (tid < o) red[tid] += red[tid + o]; __syncthreads(); }
  float w = e / red[0];
  vw[n * SSAMP + tid] = w + (1.0f - w);    // reference: vw + stop_grad(1 - vw)
}

// --------------------------- gather + bf16 pack ----------------------------
__global__ __launch_bounds__(256)
void gather_rows(const float* __restrict__ feats, const float* __restrict__ pos,
                 const int* __restrict__ feat_idx, const int* __restrict__ batch_idx,
                 const float* __restrict__ vw,
                 __bf16* __restrict__ keysb, __bf16* __restrict__ valsb) {
  const int row = blockIdx.x * 4 + (threadIdx.x >> 6);
  const int t   = threadIdx.x & 63;
  const int n   = row >> 8;
  const int fi  = feat_idx[row];
  const int b   = batch_idx[n];
  const float w = vw[row];
  const float4* fr = (const float4*)(feats + ((size_t)fi * BATCH + b) * DIM);
  const float4* pr = (const float4*)(pos   + ((size_t)fi * BATCH + b) * DIM);
  float4 f = fr[t];
  float4 p = pr[t];
  __bf16* kd = keysb + (size_t)row * DIM + t * 4;
  __bf16* vd = valsb + (size_t)row * DIM + t * 4;
  kd[0] = (__bf16)(f.x + p.x); kd[1] = (__bf16)(f.y + p.y);
  kd[2] = (__bf16)(f.z + p.z); kd[3] = (__bf16)(f.w + p.w);
  vd[0] = (__bf16)(w * f.x);   vd[1] = (__bf16)(w * f.y);
  vd[2] = (__bf16)(w * f.z);   vd[3] = (__bf16)(w * f.w);
}

// --------------------------- WMMA bf16 GEMM --------------------------------
// C[M,Nout] = A[M,256] @ W[Nout,256]^T + bias.
// Block = 8 waves, block tile = 128 rows x NT*16 cols.  A slab (128x32 bf16,
// 8KB) staged to LDS via CDNA5 global_load_async_to_lds_b128, double-buffered.
// Fragment layouts (A/B use contiguous K, not the ISA table order):
//   A 16x32 bf16 : lane row = lane&15, K = (lane>>4)*16 + e (16 contiguous)
//   B 32x16 bf16 : lane col = lane&15, K = (lane>>4)*16 + e (16 contiguous)
//   C 16x16 f32  : elem j -> row = 8*(lane>>4)+j, col = lane&15
template <int NT>
__global__ __launch_bounds__(256)
void gemm_bf16_wmma(const __bf16* __restrict__ A, const __bf16* __restrict__ W,
                    const float* __restrict__ bias, __bf16* __restrict__ C,
                    int M, int Nout) {
  __shared__ __align__(16) __bf16 atile[2][128 * 32];   // 2 x 8KB
  const int tid  = threadIdx.x;
  const int lane = tid & 31;
  const int wave = tid >> 5;
  const int blkM = blockIdx.x * 128;
  const int tn   = blockIdx.y * (NT * 16);
  // async-copy mapping: thread -> (row tid>>2 and +64, 16B quarter tid&3)
  const int crow = tid >> 2;
  const int cq   = tid & 3;
  const __bf16* gsrc0 = A + (size_t)(blkM + crow) * DIM + cq * 8;
  const __bf16* gsrc1 = gsrc0 + (size_t)64 * DIM;

  auto issue_copy = [&](int kk, int buf) {
    unsigned l0 = (unsigned)(uintptr_t)&atile[buf][crow * 32 + cq * 8];
    unsigned l1 = (unsigned)(uintptr_t)&atile[buf][(crow + 64) * 32 + cq * 8];
    asm volatile("global_load_async_to_lds_b128 %0, %1, off"
                 :: "v"(l0), "v"((ull)(uintptr_t)(gsrc0 + kk)) : "memory");
    asm volatile("global_load_async_to_lds_b128 %0, %1, off"
                 :: "v"(l1), "v"((ull)(uintptr_t)(gsrc1 + kk)) : "memory");
  };

  issue_copy(0, 0);

  v8f acc[NT] = {};
  const int ka   = (lane >> 4) * 16;
  const int kb   = (lane >> 4) * 16;
  const int arow = wave * 16 + (lane & 15);

  for (int step = 0; step < DIM / 32; ++step) {
    const int kk = step * 32;
    asm volatile("s_wait_asynccnt 0x0" ::: "memory");
    __syncthreads();                      // all waves' slabs visible in LDS
    if (step + 1 < DIM / 32) issue_copy(kk + 32, (step + 1) & 1);

    const __bf16* ap = &atile[step & 1][arow * 32];
    v16bf af;
    #pragma unroll
    for (int e = 0; e < 8; ++e) af[e]     = ap[ka + e];
    #pragma unroll
    for (int e = 0; e < 8; ++e) af[8 + e] = ap[ka + 8 + e];
    #pragma unroll
    for (int t = 0; t < NT; ++t) {
      const int c = tn + t * 16 + (lane & 15);
      const __bf16* Wp = W + (size_t)c * DIM + kk + kb;
      v16bf bf;
      #pragma unroll
      for (int e = 0; e < 16; ++e) bf[e] = Wp[e];
      acc[t] = __builtin_amdgcn_wmma_f32_16x16x32_bf16(
                 false, af, false, bf, (short)0, acc[t], false, false);
    }
  }

  const int rb = blkM + wave * 16 + (lane >> 4) * 8;
  #pragma unroll
  for (int t = 0; t < NT; ++t) {
    const int c = tn + t * 16 + (lane & 15);
    const float bv = bias[c];
    #pragma unroll
    for (int j = 0; j < 8; ++j)
      C[(size_t)(rb + j) * Nout + c] = (__bf16)(acc[t][j] + bv);
  }
}

// --------------------------- attention per slot ----------------------------
__global__ __launch_bounds__(256)
void attn_kernel(const __bf16* __restrict__ q, const __bf16* __restrict__ k,
                 const __bf16* __restrict__ v, __bf16* __restrict__ ctx) {
  __shared__ float qs[DIM];
  __shared__ float sc[NHEAD * SSAMP];
  const int n = blockIdx.x, tid = threadIdx.x;
  qs[tid] = (float)q[(size_t)n * DIM + tid];
  __syncthreads();
  {
    const int s = tid;
    const __bf16* krow = k + ((size_t)n * SSAMP + s) * DIM;
    #pragma unroll
    for (int h = 0; h < NHEAD; ++h) {
      float d = 0.f;
      #pragma unroll 8
      for (int i = 0; i < HD_; ++i) d += qs[h * HD_ + i] * (float)krow[h * HD_ + i];
      sc[h * SSAMP + s] = d * 0.17677669529663687f;   // 1/sqrt(32)
    }
  }
  __syncthreads();
  {
    const int h = tid >> 5, ln = tid & 31;
    float vals[8], mx = -3.4e38f;
    #pragma unroll
    for (int i = 0; i < 8; ++i) { vals[i] = sc[h * SSAMP + ln + 32 * i]; mx = fmaxf(mx, vals[i]); }
    for (int o = 16; o > 0; o >>= 1) mx = fmaxf(mx, __shfl_xor(mx, o, 32));
    float sum = 0.f;
    #pragma unroll
    for (int i = 0; i < 8; ++i) { vals[i] = __expf(vals[i] - mx); sum += vals[i]; }
    for (int o = 16; o > 0; o >>= 1) sum += __shfl_xor(sum, o, 32);
    float inv = 1.f / sum;
    #pragma unroll
    for (int i = 0; i < 8; ++i) sc[h * SSAMP + ln + 32 * i] = vals[i] * inv;
  }
  __syncthreads();
  {
    const int h = tid >> 5, d = tid & 31;
    float c = 0.f;
    for (int s = 0; s < SSAMP; ++s)
      c += sc[h * SSAMP + s] * (float)v[((size_t)n * SSAMP + s) * DIM + h * HD_ + d];
    ctx[(size_t)n * DIM + h * HD_ + d] = (__bf16)c;
  }
}

// --------------------------- residual + LayerNorm --------------------------
__global__ __launch_bounds__(256)
void slot_ln(const float* __restrict__ slots_in, const __bf16* __restrict__ delta,
             const float* __restrict__ g, const float* __restrict__ bta,
             float* __restrict__ out_slots, __bf16* __restrict__ lnb16) {
  __shared__ float r1[256], r2[256];
  const int n = blockIdx.x, d = threadIdx.x;
  float x = slots_in[(size_t)n * DIM + d] + (float)delta[(size_t)n * DIM + d];
  r1[d] = x; r2[d] = x * x; __syncthreads();
  for (int o = 128; o > 0; o >>= 1) { if (d < o) { r1[d] += r1[d + o]; r2[d] += r2[d + o]; } __syncthreads(); }
  float mu  = r1[0] * (1.0f / DIM);
  float var = r2[0] * (1.0f / DIM) - mu * mu;
  float y = (x - mu) * rsqrtf(var + 1e-5f) * g[d] + bta[d];
  out_slots[(size_t)n * DIM + d] = y;
  lnb16[(size_t)n * DIM + d] = (__bf16)y;
}

// --------------------------- seg softmax + scatter + spread ----------------
__global__ __launch_bounds__(256)
void seg_scatter(const __bf16* __restrict__ ps, const __bf16* __restrict__ pf,
                 const int* __restrict__ feat_idx,
                 const float* __restrict__ cwg, const float* __restrict__ ckern,
                 float* __restrict__ seg_out, float* __restrict__ curio_out) {
  __shared__ float    psl[192];
  __shared__ unsigned bm[HW_ / 32];     // occupancy bitmap of sample points (2KB)
  __shared__ float    kern[9], cw[3];
  const int n = blockIdx.x, tid = threadIdx.x;
  if (tid < 192) psl[tid] = (float)ps[(size_t)n * 192 + tid];
  bm[tid] = 0u;
  bm[tid + 256] = 0u;
  if (tid < 9) kern[tid] = ckern[tid];
  if (tid < 3) cw[tid]   = cwg[tid];
  __syncthreads();
  const int fi = feat_idx[n * SSAMP + tid];
  atomicOr(&bm[fi >> 5], 1u << (fi & 31));
  const __bf16* pfr = pf + ((size_t)n * SSAMP + tid) * 192;
  float l[3];
  #pragma unroll
  for (int kk = 0; kk < 3; ++kk) {
    float s = 0.f;
    #pragma unroll 8
    for (int dd = 0; dd < 64; ++dd) s += psl[kk * 64 + dd] * (float)pfr[kk * 64 + dd];
    l[kk] = s;
  }
  float m  = fmaxf(l[0], fmaxf(l[1], l[2]));
  float e0 = __expf(l[0] - m), e1 = __expf(l[1] - m), e2 = __expf(l[2] - m);
  float inv = 1.f / (e0 + e1 + e2);
  float p0 = e0 * inv, p1 = e1 * inv, p2 = e2 * inv;
  float* segn = seg_out + (size_t)n * 3 * HW_;
  segn[fi] = p0; segn[HW_ + fi] = p1; segn[2 * HW_ + fi] = p2;
  float samp = cw[0] * p0 + cw[1] * p1 + cw[2] * p2;
  float* cmo = curio_out + (size_t)n * HW_;
  cmo[fi] = samp;                       // .set at sample points
  __syncthreads();                      // bitmap complete before spread
  const int y = fi >> 7, x = fi & 127;
  #pragma unroll
  for (int dy = -1; dy <= 1; ++dy) {
    #pragma unroll
    for (int dx = -1; dx <= 1; ++dx) {
      if (dy == 0 && dx == 0) continue;
      int y2 = y + dy, x2 = x + dx;
      if (y2 < 0 || y2 >= HH_ || x2 < 0 || x2 >= WW_) continue;
      int fi2 = (y2 << 7) | x2;
      if ((bm[fi2 >> 5] >> (fi2 & 31)) & 1u) continue;   // spread zeroed at samples
      atomicAdd(&cmo[fi2], samp * kern[(dy + 1) * 3 + (dx + 1)]);
    }
  }
}

// ---------------------------------------------------------------------------
extern "C" void kernel_launch(void* const* d_in, const int* in_sizes, int n_in,
                              void* d_out, int out_size, void* d_ws, size_t ws_size,
                              hipStream_t stream) {
  (void)in_sizes; (void)n_in; (void)out_size; (void)ws_size;
  const float* features = (const float*)d_in[0];
  const float* pos      = (const float*)d_in[1];
  const float* slots    = (const float*)d_in[2];
  const float* seg_in   = (const float*)d_in[3];
  const float* curio_in = (const float*)d_in[4];
  const int*   batch_ix = (const int*)d_in[5];
  const int*   cov_sel  = (const int*)d_in[6];
  const float* inpw     = (const float*)d_in[8];
  const float* inpb     = (const float*)d_in[9];
  const float* outw     = (const float*)d_in[10];
  const float* outb     = (const float*)d_in[11];
  const float* lng      = (const float*)d_in[12];
  const float* lnbv     = (const float*)d_in[13];
  const float* slotw    = (const float*)d_in[14];
  const float* slotb    = (const float*)d_in[15];
  const float* featw    = (const float*)d_in[16];
  const float* featb    = (const float*)d_in[17];
  const float* cwg      = (const float*)d_in[18];
  const float* ckern    = (const float*)d_in[19];

  float* out_slots = (float*)d_out;                               // 2048*256
  float* seg_out   = out_slots + (size_t)NSLOT * DIM;             // 2048*3*16384
  float* curio_out = seg_out + (size_t)NSLOT * 3 * HW_;           // 2048*16384

  const size_t NS = (size_t)NSLOT * SSAMP;                        // 524288 rows

  char* base = (char*)d_ws;
  size_t off = 0;
  auto alloc = [&](size_t bytes) -> void* {
    void* p = base + off;
    off += (bytes + 255) & ~(size_t)255;
    return p;
  };
  int*    feat_idx_w = (int*)   alloc(NS * sizeof(int));
  float*  s_cur_w    = (float*) alloc(NS * sizeof(float));
  float*  vw_w       = (float*) alloc(NS * sizeof(float));
  __bf16* keysb      = (__bf16*)alloc(NS * DIM * 2);      // 268 MB
  __bf16* valsb      = (__bf16*)alloc(NS * DIM * 2);      // 268 MB
  __bf16* kbuf       = (__bf16*)alloc(NS * DIM * 2);      // 268 MB
  __bf16* vbuf       = (__bf16*)alloc(NS * DIM * 2);      // 268 MB
  __bf16* pfb        = (__bf16*)alloc(NS * 192 * 2);      // 201 MB
  __bf16* qb         = (__bf16*)alloc((size_t)NSLOT * DIM * 2);
  __bf16* slotsb     = (__bf16*)alloc((size_t)NSLOT * DIM * 2);
  __bf16* ctxb       = (__bf16*)alloc((size_t)NSLOT * DIM * 2);
  __bf16* deltab     = (__bf16*)alloc((size_t)NSLOT * DIM * 2);
  __bf16* slotslnb   = (__bf16*)alloc((size_t)NSLOT * DIM * 2);
  __bf16* psb        = (__bf16*)alloc((size_t)NSLOT * 192 * 2);
  __bf16* wqkvb      = (__bf16*)alloc((size_t)768 * DIM * 2);
  __bf16* woutb      = (__bf16*)alloc((size_t)DIM * DIM * 2);
  __bf16* wslotb     = (__bf16*)alloc((size_t)192 * DIM * 2);
  __bf16* wfeatb     = (__bf16*)alloc((size_t)192 * DIM * 2);

  // 0) initialize outputs (reference copies its input seg/curio maps)
  hipMemcpyAsync(seg_out, seg_in, (size_t)NSLOT * 3 * HW_ * sizeof(float),
                 hipMemcpyDeviceToDevice, stream);
  hipMemcpyAsync(curio_out, curio_in, (size_t)NSLOT * HW_ * sizeof(float),
                 hipMemcpyDeviceToDevice, stream);

  // 1) weight / activation conversions to bf16
  auto cvt = [&](const float* s, __bf16* d, int cnt) {
    cvt_f32_bf16<<<(cnt + 255) / 256, 256, 0, stream>>>(s, d, cnt);
  };
  cvt(inpw,  wqkvb,  768 * DIM);
  cvt(outw,  woutb,  DIM * DIM);
  cvt(slotw, wslotb, 192 * DIM);
  cvt(featw, wfeatb, 192 * DIM);
  cvt(slots, slotsb, NSLOT * DIM);

  // 2) full per-slot sort (hybrid bitonic, 128KB dynamic LDS)
  topk_sort<<<NSLOT, 1024, HW_ * sizeof(ull), stream>>>(
      curio_in, cov_sel, feat_idx_w, s_cur_w);

  // 3) value-weight softmax
  vw_softmax<<<NSLOT, 256, 0, stream>>>(s_cur_w, vw_w);

  // 4) gather -> bf16 keys / values
  gather_rows<<<(int)(NS / 4), 256, 0, stream>>>(
      features, pos, feat_idx_w, batch_ix, vw_w, keysb, valsb);

  // 5) big WMMA GEMMs (NT=8 -> 128-wide tiles for Nout=256; NT=6 -> 96 for 192)
  const __bf16* wq = wqkvb;
  const __bf16* wk = wqkvb + (size_t)256 * DIM;
  const __bf16* wv = wqkvb + (size_t)512 * DIM;
  gemm_bf16_wmma<8><<<dim3(NSLOT / 128, 2), 256, 0, stream>>>(slotsb, wq, inpb,        qb,   NSLOT, DIM);
  gemm_bf16_wmma<8><<<dim3((int)(NS / 128), 2), 256, 0, stream>>>(keysb, wk, inpb + 256, kbuf, (int)NS, DIM);
  gemm_bf16_wmma<8><<<dim3((int)(NS / 128), 2), 256, 0, stream>>>(valsb, wv, inpb + 512, vbuf, (int)NS, DIM);
  gemm_bf16_wmma<6><<<dim3((int)(NS / 128), 2), 256, 0, stream>>>(keysb, wfeatb, featb, pfb,  (int)NS, 192);

  // 6) attention
  attn_kernel<<<NSLOT, 256, 0, stream>>>(qb, kbuf, vbuf, ctxb);

  // 7) output projection, residual + LN
  gemm_bf16_wmma<8><<<dim3(NSLOT / 128, 2), 256, 0, stream>>>(ctxb, woutb, outb, deltab, NSLOT, DIM);
  slot_ln<<<NSLOT, 256, 0, stream>>>(slots, deltab, lng, lnbv, out_slots, slotslnb);

  // 8) ps projection + seg scatter / curiosity spread
  gemm_bf16_wmma<6><<<dim3(NSLOT / 128, 2), 256, 0, stream>>>(slotslnb, wslotb, slotb, psb, NSLOT, 192);
  seg_scatter<<<NSLOT, 256, 0, stream>>>(psb, pfb, feat_idx_w, cwg, ckern, seg_out, curio_out);
}
